// Codebook_9414568313012
// MI455X (gfx1250) — matrix operand order repaired and checked by reading, not compiled, not run on hardware
//
#include <hip/hip_runtime.h>
#include <hip/hip_bf16.h>

typedef __attribute__((ext_vector_type(16))) __bf16 v16bf;
typedef __attribute__((ext_vector_type(8)))  float  v8f;

#define NPIX   65536      // 16 * 64 * 64
#define CCH    256
#define KCODE  1024
#define HWSZ   4096       // 64*64
#define IMSTR  1048576    // 256*4096 floats per image

// ---------------- Kernel 0: codebook fp32 -> bf16, per-code squared norm ----
__global__ __launch_bounds__(256) void prep_kernel(const float* __restrict__ w,
                                                   unsigned short* __restrict__ wbf,
                                                   float* __restrict__ wnorm) {
    int k = blockIdx.x, t = threadIdx.x;
    float v = w[k * CCH + t];
    __bf16 bv = (__bf16)v;
    wbf[k * CCH + t] = *reinterpret_cast<unsigned short*>(&bv);
    __shared__ float red[256];
    red[t] = v * v;
    __syncthreads();
    for (int off = 128; off > 0; off >>= 1) {
        if (t < off) red[t] += red[t + off];
        __syncthreads();
    }
    if (t == 0) wnorm[k] = red[0];
}

// ---------------- Kernel 1: WMMA distance GEMM + argmin ---------------------
// One wave per 32 pixels (two 16x16 M-tiles). dist = ||w||^2 - 2 x.w^T
__global__ __launch_bounds__(256) void argmin_kernel(const float* __restrict__ x,
                                                     const unsigned short* __restrict__ wbf_us,
                                                     const float* __restrict__ wnorm,
                                                     int* __restrict__ idx_out) {
    const __bf16* __restrict__ wbf = reinterpret_cast<const __bf16*>(wbf_us);
    const int lane  = threadIdx.x & 31;
    const int wave  = (int)((blockIdx.x * blockDim.x + threadIdx.x) >> 5); // 0..2047
    const int g     = lane >> 4;   // lane half
    const int m     = lane & 15;   // A row / B-D column
    const int tile0 = wave * 32;

    // ---- Load A tiles (16 pixels x 32 channels, bf16) in contiguous k order, same as B:
    // lane half g, element j -> K = j+16g
    v16bf A[2][8];
    #pragma unroll
    for (int mt = 0; mt < 2; ++mt) {
        int p  = tile0 + mt * 16 + m;
        int b  = p >> 12;
        int hw = p & (HWSZ - 1);
        const float* xb = x + (size_t)b * IMSTR + hw;
        #pragma unroll
        for (int cc = 0; cc < 8; ++cc) {
            #pragma unroll
            for (int j = 0; j < 16; ++j) {
                int kk = j + 16 * g;
                A[mt][cc][j] = (__bf16)xb[(size_t)(cc * 32 + kk) * HWSZ];
            }
        }
    }

    float minv[2][8];
    int   mini[2][8];
    #pragma unroll
    for (int mt = 0; mt < 2; ++mt)
        #pragma unroll
        for (int r = 0; r < 8; ++r) { minv[mt][r] = 3.4e38f; mini[mt][r] = 0; }

    for (int ct = 0; ct < 64; ++ct) {
        const int code = ct * 16 + m;                 // code column owned by this lane
        const __bf16* wb = wbf + (size_t)code * CCH + g * 16;
        v8f acc0 = {}, acc1 = {};
        #pragma unroll
        for (int cc = 0; cc < 8; ++cc) {
            v16bf B = *(const v16bf*)(wb + cc * 32);  // 32B contiguous, 32B aligned
            acc0 = __builtin_amdgcn_wmma_f32_16x16x32_bf16(false, A[0][cc], false, B,
                                                           (short)0, acc0, false, false);
            acc1 = __builtin_amdgcn_wmma_f32_16x16x32_bf16(false, A[1][cc], false, B,
                                                           (short)0, acc1, false, false);
        }
        float wn = wnorm[code];
        #pragma unroll
        for (int r = 0; r < 8; ++r) {
            float d0 = wn - 2.0f * acc0[r];
            if (d0 < minv[0][r]) { minv[0][r] = d0; mini[0][r] = code; }
            float d1 = wn - 2.0f * acc1[r];
            if (d1 < minv[1][r]) { minv[1][r] = d1; mini[1][r] = code; }
        }
    }

    // ---- argmin across the 16 code lanes (xor masks stay inside the 16-group)
    #pragma unroll
    for (int mt = 0; mt < 2; ++mt) {
        #pragma unroll
        for (int r = 0; r < 8; ++r) {
            float v = minv[mt][r];
            int   i = mini[mt][r];
            #pragma unroll
            for (int off = 1; off < 16; off <<= 1) {
                float ov = __shfl_xor(v, off, 32);
                int   oi = __shfl_xor(i, off, 32);
                if (ov < v || (ov == v && oi < i)) { v = ov; i = oi; }
            }
            if (m == 0) idx_out[tile0 + mt * 16 + g * 8 + r] = i;  // lanes 0 and 16
        }
    }
}

// ---------------- Kernel 2: zq scatter (NCHW) + exact fp32 loss partials ----
__global__ __launch_bounds__(256) void zq_loss_kernel(const float* __restrict__ x,
                                                      const float* __restrict__ w,
                                                      const int* __restrict__ idx,
                                                      float* __restrict__ zq,
                                                      float* __restrict__ seq,
                                                      float* __restrict__ partial) {
    int t = threadIdx.x;
    int p = blockIdx.x * 256 + t;
    int k = idx[p];
    seq[p] = (float)k;
    int b  = p >> 12;
    int hw = p & (HWSZ - 1);
    const float* wrow = w + (size_t)k * CCH;
    size_t base = (size_t)b * IMSTR + hw;
    float s = 0.0f;
    for (int c = 0; c < CCH; ++c) {
        float wv = wrow[c];                  // gather, L2-resident (1 MB codebook)
        float xv = x[base + (size_t)c * HWSZ];
        zq[base + (size_t)c * HWSZ] = wv;    // straight-through: output == w[k] numerically
        float d = wv - xv;
        s += d * d;
    }
    __shared__ float red[256];
    red[t] = s;
    __syncthreads();
    for (int off = 128; off > 0; off >>= 1) {
        if (t < off) red[t] += red[t + off];
        __syncthreads();
    }
    if (t == 0) partial[blockIdx.x] = red[0];
}

// ---------------- Kernel 3: deterministic final loss reduction --------------
__global__ __launch_bounds__(256) void finalize_kernel(const float* __restrict__ partial,
                                                       float* __restrict__ losses) {
    __shared__ float red[256];
    int t = threadIdx.x;
    red[t] = partial[t];
    __syncthreads();
    for (int off = 128; off > 0; off >>= 1) {
        if (t < off) red[t] += red[t + off];
        __syncthreads();
    }
    if (t == 0) {
        float m = red[0] / 16777216.0f;      // mean over N*C
        losses[0] = m;                       // codebook_loss
        losses[1] = m;                       // commitment_loss (same value)
    }
}

extern "C" void kernel_launch(void* const* d_in, const int* in_sizes, int n_in,
                              void* d_out, int out_size, void* d_ws, size_t ws_size,
                              hipStream_t stream) {
    const float* x = (const float*)d_in[0];    // (16, 256, 64, 64)
    const float* w = (const float*)d_in[1];    // (1024, 256)

    float* zq     = (float*)d_out;                      // 16777216 elems
    float* seq    = (float*)d_out + 16777216;           // 65536 elems
    float* losses = (float*)d_out + 16842752;           // 2 elems

    char* ws = (char*)d_ws;
    unsigned short* wbf   = (unsigned short*)(ws);                 // 512 KB
    float*          wnorm = (float*)(ws + 524288);                 // 4 KB
    int*            idxb  = (int*)(ws + 528384);                   // 256 KB
    float*          part  = (float*)(ws + 790528);                 // 1 KB

    prep_kernel<<<KCODE, 256, 0, stream>>>(w, wbf, wnorm);
    argmin_kernel<<<NPIX / 256, 256, 0, stream>>>(x, wbf, wnorm, idxb);  // 8 waves/block
    zq_loss_kernel<<<NPIX / 256, 256, 0, stream>>>(x, w, idxb, zq, seq, part);
    finalize_kernel<<<1, 256, 0, stream>>>(part, losses);
}
